// GCN_38019050504947
// MI455X (gfx1250) — compile-verified, matrix operand order checked by reading, not run on hardware
//
#include <hip/hip_runtime.h>
#include <hip/hip_bf16.h>

// ---------------- problem constants ----------------
#define N_CELL   32768
#define N_TISSUE 2048
#define D_FEAT   512
#define D_HID    256
#define N_TOK    3
#define N_HEAD   8
#define HD       32          // head dim
#define CLASSES  7
#define EPSV     1e-5f
#define LSLOPE   0.01f

typedef _Float16 v16h __attribute__((ext_vector_type(16)));
typedef _Float16 h8   __attribute__((ext_vector_type(8)));
typedef float    v8f  __attribute__((ext_vector_type(8)));
typedef float    f4   __attribute__((ext_vector_type(4)));

// =====================================================================
// WMMA GEMM:  C[M,N] = act( A0@W0 (+ A1@W1) + bias )
//   A:  fp32 row-major, leading dim lda (lda % 8 == 0), M % 64 == 0
//   Wt: f16, PRE-TRANSPOSED to N x K row-major, K % 32 == 0, N % 64 == 0
//   ACT: 0 = none, 1 = relu, 2 = leaky-relu(0.01)   (compile-time)
//   DUAL: fused second A@W pair sharing the accumulators (compile-time)
// 128 threads = 4 waves; wave computes 16x64 (4 accumulators) -> 4 (8 dual)
// v_wmma per k-step per wave; A fragment reused across all 4 N tiles.
// B fragments load straight from global f16 (two b128 per tile) - weights
// are small and L2/L0 resident; no LDS, no inner-loop barriers/branches.
// =====================================================================
template<bool DUAL, int ACT>
__launch_bounds__(128)
__global__ void gemm_wmma(const float* __restrict__ A0, int lda0,
                          const float* __restrict__ A1, int lda1,
                          const _Float16* __restrict__ Wt0,
                          const _Float16* __restrict__ Wt1,
                          const float* __restrict__ bias,
                          float* __restrict__ C, int ldc,
                          int M, int N, int K)
{
    const int tid  = threadIdx.x;
    const int wave = tid >> 5;
    const int lane = tid & 31;
    const int hi   = lane >> 4;      // 0 or 1
    const int lsub = lane & 15;
    const int n0   = blockIdx.y * 64;
    const int m0   = blockIdx.x * 64 + wave * 16;
    const int arow = m0 + lsub;      // A row held by this lane

    v8f acc[4] = {};

    for (int kk = 0; kk < K; kk += 32) {
        // ---- A fragment: halves e=0..7 -> K=kk+hi*8+e, e=8..15 -> +16
        const float* ap0 = A0 + (size_t)arow * lda0 + kk + hi * 8;
        f4 x0 = *(const f4*)(ap0);
        f4 x1 = *(const f4*)(ap0 + 4);
        f4 x2 = *(const f4*)(ap0 + 16);
        f4 x3 = *(const f4*)(ap0 + 20);
        if (kk + 32 < K)
            __builtin_prefetch((const void*)(ap0 + 32), 0, 1);
        v16h a0;
        #pragma unroll
        for (int j = 0; j < 4; ++j) {
            a0[j]      = (_Float16)x0[j];
            a0[4 + j]  = (_Float16)x1[j];
            a0[8 + j]  = (_Float16)x2[j];
            a0[12 + j] = (_Float16)x3[j];
        }
        v16h a1;
        if (DUAL) {
            const float* ap1 = A1 + (size_t)arow * lda1 + kk + hi * 8;
            f4 y0 = *(const f4*)(ap1);
            f4 y1 = *(const f4*)(ap1 + 4);
            f4 y2 = *(const f4*)(ap1 + 16);
            f4 y3 = *(const f4*)(ap1 + 20);
            #pragma unroll
            for (int j = 0; j < 4; ++j) {
                a1[j]      = (_Float16)y0[j];
                a1[4 + j]  = (_Float16)y1[j];
                a1[8 + j]  = (_Float16)y2[j];
                a1[12 + j] = (_Float16)y3[j];
            }
        }

        // ---- 4 N-tiles share the A fragment
        #pragma unroll
        for (int nt = 0; nt < 4; ++nt) {
            int col = n0 + nt * 16 + lsub;
            const _Float16* wp = Wt0 + (size_t)col * K + kk + hi * 8;
            h8 blo = *(const h8*)wp;
            h8 bhi = *(const h8*)(wp + 16);
            v16h b0;
            #pragma unroll
            for (int j = 0; j < 8; ++j) { b0[j] = blo[j]; b0[8 + j] = bhi[j]; }
            acc[nt] = __builtin_amdgcn_wmma_f32_16x16x32_f16(
                          false, a0, false, b0, (short)0, acc[nt], false, false);
            if (DUAL) {
                const _Float16* wq = Wt1 + (size_t)col * K + kk + hi * 8;
                h8 clo = *(const h8*)wq;
                h8 chi = *(const h8*)(wq + 16);
                v16h b1;
                #pragma unroll
                for (int j = 0; j < 8; ++j) { b1[j] = clo[j]; b1[8 + j] = chi[j]; }
                acc[nt] = __builtin_amdgcn_wmma_f32_16x16x32_f16(
                              false, a1, false, b1, (short)0, acc[nt], false, false);
            }
        }
    }

    // ---- epilogue: acc[nt][r] -> row m0 + r + hi*8, col n0 + nt*16 + lsub
    // (M % 64 == 0 at every call site, so no row bounds check needed)
    #pragma unroll
    for (int nt = 0; nt < 4; ++nt) {
        int col = n0 + nt * 16 + lsub;
        float bn = bias ? bias[col] : 0.0f;
        #pragma unroll
        for (int r = 0; r < 8; ++r) {
            int orow = m0 + r + hi * 8;
            float v = acc[nt][r] + bn;
            if (ACT == 1) v = v > 0.0f ? v : 0.0f;
            else if (ACT == 2) v = v > 0.0f ? v : LSLOPE * v;
            C[(size_t)orow * ldc + col] = v;
        }
    }
}

// transpose + convert weight: W (KxN f32, row-major) -> Wt (NxK f16, row-major)
__global__ void w_transpose_f16(const float* __restrict__ W, _Float16* __restrict__ Wt,
                                int K, int N)
{
    long t = (long)blockIdx.x * blockDim.x + threadIdx.x;
    if (t >= (long)K * N) return;
    int n = (int)(t / K);
    int k = (int)(t % K);
    Wt[t] = (_Float16)W[(size_t)k * N + n];
}

// ---------------- small utility kernels ----------------
__global__ void zerof(float* p, long n) {
    long t = (long)blockIdx.x * blockDim.x + threadIdx.x;
    if (t < n) p[t] = 0.0f;
}

// scatter-add x[src] into agg[dst], and count degree (on f==0 lane)
__global__ void sage_edge(const int* __restrict__ ei, int E,
                          const float* __restrict__ x,
                          float* __restrict__ agg, float* __restrict__ deg, int D)
{
    long t = (long)blockIdx.x * blockDim.x + threadIdx.x;
    long total = (long)E * D;
    if (t >= total) return;
    int e = (int)(t / D);
    int f = (int)(t % D);
    int src = ei[e];
    int dst = ei[E + e];
    atomicAdd(&agg[(size_t)dst * D + f], x[(size_t)src * D + f]);
    if (f == 0) atomicAdd(&deg[dst], 1.0f);
}

__global__ void rowdiv(float* agg, const float* deg, int M, int D) {
    long t = (long)blockIdx.x * blockDim.x + threadIdx.x;
    if (t >= (long)M * D) return;
    agg[t] /= fmaxf(deg[t / D], 1.0f);
}

// partial column sum / sumsq (<=256 cols), 128 rows per block
__global__ void col_accum(const float* __restrict__ x, int lda, int M, int N,
                          float* __restrict__ sum, float* __restrict__ sumsq)
{
    int col = threadIdx.x;
    if (col >= N) return;
    int r0 = blockIdx.x * 128;
    float s = 0.0f, s2 = 0.0f;
    for (int r = 0; r < 128; ++r) {
        int row = r0 + r;
        if (row >= M) break;
        float v = x[(size_t)row * lda + col];
        s += v; s2 += v * v;
    }
    atomicAdd(&sum[col], s);
    atomicAdd(&sumsq[col], s2);
}

// graph-norm + leaky-relu in place
__global__ void gnorm_apply(float* x, int M, int N,
                            const float* __restrict__ w, const float* __restrict__ b,
                            const float* __restrict__ ms,
                            const float* __restrict__ sum, const float* __restrict__ sumsq)
{
    long t = (long)blockIdx.x * blockDim.x + threadIdx.x;
    if (t >= (long)M * N) return;
    int col = (int)(t % N);
    float m   = sum[col] / (float)M;
    float ex2 = sumsq[col] / (float)M;
    float msv = ms[col];
    float var = ex2 - 2.0f * msv * m * m + msv * msv * m * m;
    float c   = x[t] - msv * m;
    float y   = w[col] * c * rsqrtf(var + EPSV) + b[col];
    x[t] = y > 0.0f ? y : LSLOPE * y;
}

// tokens: X[cell, slot, f] = {cls, xc, xt[assign]} + pos
__global__ void build_tokens(const float* __restrict__ xc, const float* __restrict__ xt,
                             const int* __restrict__ assign,
                             const float* __restrict__ cls, const float* __restrict__ pos,
                             float* __restrict__ X)
{
    long t = (long)blockIdx.x * blockDim.x + threadIdx.x;
    if (t >= (long)N_CELL * N_TOK * D_HID) return;
    int cell = (int)(t / (N_TOK * D_HID));
    int r    = (int)(t % (N_TOK * D_HID));
    int slot = r / D_HID;
    int f    = r % D_HID;
    float v;
    if (slot == 0)      v = cls[f];
    else if (slot == 1) v = xc[(size_t)cell * D_HID + f];
    else                v = xt[(size_t)assign[cell] * D_HID + f];
    X[t] = v + pos[slot * D_HID + f];
}

// 3x3 attention: one wave32 per (cell, head), lanes span head dim (coalesced)
__global__ void attn3(const float* __restrict__ Q, const float* __restrict__ K,
                      const float* __restrict__ V, float* __restrict__ O)
{
    int gid  = blockIdx.x * blockDim.x + threadIdx.x;
    int w    = gid >> 5;            // (cell * 8 + head)
    int lane = threadIdx.x & 31;    // d within head
    if (w >= N_CELL * N_HEAD) return;
    int cell = w >> 3;
    int h    = w & 7;
    size_t base = (size_t)cell * N_TOK * D_HID + h * HD + lane;

    float q0 = Q[base], q1 = Q[base + D_HID], q2 = Q[base + 2 * D_HID];
    float k0 = K[base], k1 = K[base + D_HID], k2 = K[base + 2 * D_HID];
    float v0 = V[base], v1 = V[base + D_HID], v2 = V[base + 2 * D_HID];

    float p[9] = { q0*k0, q0*k1, q0*k2, q1*k0, q1*k1, q1*k2, q2*k0, q2*k1, q2*k2 };
    #pragma unroll
    for (int j = 0; j < 9; ++j) {
        #pragma unroll
        for (int off = 16; off; off >>= 1) p[j] += __shfl_xor(p[j], off, 32);
    }
    const float scale = 0.17677669529663687f; // 1/sqrt(32)
    float o[3];
    #pragma unroll
    for (int q = 0; q < 3; ++q) {
        float s0 = p[q*3+0]*scale, s1 = p[q*3+1]*scale, s2 = p[q*3+2]*scale;
        float mx = fmaxf(s0, fmaxf(s1, s2));
        float e0 = expf(s0 - mx), e1 = expf(s1 - mx), e2 = expf(s2 - mx);
        float inv = 1.0f / (e0 + e1 + e2);
        o[q] = (e0 * v0 + e1 * v1 + e2 * v2) * inv;
    }
    O[base]             = o[0];
    O[base + D_HID]     = o[1];
    O[base + 2 * D_HID] = o[2];
}

// x = LN(x + r) row-wise over 256, one wave32 per row
__global__ void add_ln(float* __restrict__ x, const float* __restrict__ r,
                       const float* __restrict__ w, const float* __restrict__ b,
                       int rows)
{
    int gid  = blockIdx.x * blockDim.x + threadIdx.x;
    int wid  = gid >> 5;
    int lane = threadIdx.x & 31;
    if (wid >= rows) return;
    size_t base = (size_t)wid * D_HID;
    float v[8];
    float s = 0.0f;
    #pragma unroll
    for (int i = 0; i < 8; ++i) {
        int c = lane + i * 32;
        v[i] = x[base + c] + r[base + c];
        s += v[i];
    }
    #pragma unroll
    for (int off = 16; off; off >>= 1) s += __shfl_xor(s, off, 32);
    float m = s * (1.0f / D_HID);
    float q = 0.0f;
    #pragma unroll
    for (int i = 0; i < 8; ++i) { float d = v[i] - m; q += d * d; }
    #pragma unroll
    for (int off = 16; off; off >>= 1) q += __shfl_xor(q, off, 32);
    float inv = rsqrtf(q * (1.0f / D_HID) + EPSV);
    #pragma unroll
    for (int i = 0; i < 8; ++i) {
        int c = lane + i * 32;
        x[base + c] = (v[i] - m) * inv * w[c] + b[c];
    }
}

// att_raw[row] = T[row,:] . A2 + a2   (T is 32768 x 128)
__global__ void att_dot(const float* __restrict__ T, const float* __restrict__ A2,
                        const float* __restrict__ a2, float* __restrict__ out)
{
    int row = blockIdx.x * blockDim.x + threadIdx.x;
    if (row >= N_CELL) return;
    float s = a2[0];
    for (int k = 0; k < 128; ++k) s += T[(size_t)row * 128 + k] * A2[k];
    out[row] = s;
}

// softmax over axis 0 (single block)
__global__ void softmax_col(const float* __restrict__ x, int n, float* __restrict__ out)
{
    __shared__ float red[256];
    int t = threadIdx.x;
    float mx = -3.4e38f;
    for (int i = t; i < n; i += 256) mx = fmaxf(mx, x[i]);
    red[t] = mx; __syncthreads();
    for (int s = 128; s; s >>= 1) { if (t < s) red[t] = fmaxf(red[t], red[t + s]); __syncthreads(); }
    mx = red[0]; __syncthreads();
    float sm = 0.0f;
    for (int i = t; i < n; i += 256) sm += expf(x[i] - mx);
    red[t] = sm; __syncthreads();
    for (int s = 128; s; s >>= 1) { if (t < s) red[t] += red[t + s]; __syncthreads(); }
    sm = red[0];
    float inv = 1.0f / sm;
    for (int i = t; i < n; i += 256) out[i] = expf(x[i] - mx) * inv;
}

// head: g = mean rows -> lrelu(g@L1+l1) -> LN -> @L2 + l2 (single block)
__global__ void head_tail(const float* __restrict__ colsum,
                          const float* __restrict__ L1, const float* __restrict__ l1,
                          const float* __restrict__ lnw, const float* __restrict__ lnb,
                          const float* __restrict__ L2, const float* __restrict__ l2,
                          float* __restrict__ logits)
{
    __shared__ float g[256];
    __shared__ float g1[128];
    __shared__ float mv[2];
    int t = threadIdx.x;
    if (t < 256) g[t] = colsum[t] * (1.0f / (float)N_CELL);
    __syncthreads();
    if (t < 128) {
        float s = l1[t];
        for (int k = 0; k < 256; ++k) s += g[k] * L1[k * 128 + t];
        g1[t] = s > 0.0f ? s : LSLOPE * s;
    }
    __syncthreads();
    if (t == 0) {
        float m = 0.0f;
        for (int k = 0; k < 128; ++k) m += g1[k];
        m *= (1.0f / 128.0f);
        float v = 0.0f;
        for (int k = 0; k < 128; ++k) { float d = g1[k] - m; v += d * d; }
        v *= (1.0f / 128.0f);
        mv[0] = m; mv[1] = rsqrtf(v + EPSV);
    }
    __syncthreads();
    if (t < 128) g1[t] = (g1[t] - mv[0]) * mv[1] * lnw[t] + lnb[t];
    __syncthreads();
    if (t < CLASSES) {
        float s = l2[t];
        for (int k = 0; k < 128; ++k) s += g1[k] * L2[k * CLASSES + t];
        logits[t] = s;
    }
}

__global__ void write_out(const float* __restrict__ logits, const int* __restrict__ ei,
                          float* __restrict__ out, int E2)
{
    int i = blockIdx.x * blockDim.x + threadIdx.x;
    if (i < CLASSES) out[i] = logits[i];
    else if (i < CLASSES + E2) out[i] = (float)ei[i - CLASSES];
}

// =====================================================================
// host driver
// =====================================================================
static inline dim3 g1d(long n, int b = 256) { return dim3((unsigned)((n + b - 1) / b)); }

extern "C" void kernel_launch(void* const* d_in, const int* in_sizes, int n_in,
                              void* d_out, int out_size, void* d_ws, size_t ws_size,
                              hipStream_t stream)
{
    if (n_in < 70) return;
    const float* x_cell   = (const float*)d_in[0];
    const float* x_tissue = (const float*)d_in[1];
    const int*   eic      = (const int*)d_in[2];
    const int*   eit      = (const int*)d_in[3];
    const int*   assign   = (const int*)d_in[4];
    const int E_cell = in_sizes[2] / 2;
    const int E_tis  = in_sizes[3] / 2;
    auto P = [&](int i) { return (const float*)d_in[i]; };

    const float* gn_w  = P(23);
    const float* gn_b  = P(24);
    const float* gn_ms = P(25);

    // ---- workspace layout (floats, 256B aligned chunks) ----
    float* f = (float*)d_ws;
    size_t off = 0;
    auto alloc = [&](size_t n) { float* p = f + off; off += (n + 63) & ~(size_t)63; return p; };
    const size_t TOKN = (size_t)N_CELL * N_TOK * D_HID; // 25,165,824
    float* X   = alloc(TOKN);
    float* Qb  = alloc(TOKN);
    float* Kb  = alloc(TOKN);
    float* Vb  = alloc(TOKN);
    float* Tb  = alloc(TOKN);
    float* deg     = alloc(N_CELL);
    float* colsum  = alloc(512);
    float* colsumsq= alloc(512);
    float* attraw  = alloc(N_CELL);
    float* logitsb = alloc(16);
    // f16 region for pre-transposed weights (25 matrices, ~3.7 MB)
    float* hreg = alloc(1000000);  // 2,000,000 halves
    if (ws_size < off * sizeof(float)) return;

    // ---- pre-transpose + f16-convert all WMMA-consumed weights ----
    static const int wt_idx[25] = { 5,7, 8,10, 11,13, 14,16, 17,19, 20,22,
                                    26,28,30,32,36,38, 42,44,46,48,52,54, 60 };
    static const int wt_K[25]   = { 512,512, 256,256, 256,256, 512,512, 256,256, 256,256,
                                    256,256,256,256,256,256, 256,256,256,256,256,256, 256 };
    static const int wt_N[25]   = { 256,256, 256,256, 256,256, 256,256, 256,256, 256,256,
                                    256,256,256,256,256,256, 256,256,256,256,256,256, 128 };
    _Float16* wt_of[70] = {};
    {
        _Float16* hp = (_Float16*)hreg;
        for (int i = 0; i < 25; ++i) {
            long n = (long)wt_K[i] * wt_N[i];
            wt_of[wt_idx[i]] = hp;
            w_transpose_f16<<<g1d(n), 256, 0, stream>>>(P(wt_idx[i]), hp, wt_K[i], wt_N[i]);
            hp += (n + 127) & ~(long)127;
        }
    }

    // SAGE-phase aliases (inside regions not yet used by the transformer)
    float* aggc = Qb;                       // 32768x512
    float* hc0  = Kb;                       // 32768x256
    float* hc1  = Kb + (size_t)N_CELL * D_HID;
    float* ht0  = Vb;                       // 2048x256
    float* ht1  = Vb + (size_t)N_TISSUE * D_HID;
    float* aggt = Vb + 2 * (size_t)N_TISSUE * D_HID; // 2048x512

    float* out    = (float*)d_out;
    float* outAtt = out + CLASSES + (size_t)2 * E_cell;

    // ---- one SAGE layer + graph-norm + lrelu ----
    auto sage_layer = [&](const float* xin, int M, int Din, const int* ei, int E,
                          const _Float16* Wl, const float* bl, const _Float16* Wr,
                          float* agg, float* hout)
    {
        zerof<<<g1d((long)M * Din), 256, 0, stream>>>(agg, (long)M * Din);
        zerof<<<g1d(M), 256, 0, stream>>>(deg, M);
        sage_edge<<<g1d((long)E * Din), 256, 0, stream>>>(ei, E, xin, agg, deg, Din);
        rowdiv<<<g1d((long)M * Din), 256, 0, stream>>>(agg, deg, M, Din);
        gemm_wmma<true, 0><<<dim3(M / 64, D_HID / 64), 128, 0, stream>>>(
            agg, Din, xin, Din, Wl, Wr, bl, hout, D_HID, M, D_HID, Din);
        zerof<<<1, 256, 0, stream>>>(colsum, 256);
        zerof<<<1, 256, 0, stream>>>(colsumsq, 256);
        col_accum<<<g1d(M, 128).x, 256, 0, stream>>>(hout, D_HID, M, D_HID, colsum, colsumsq);
        gnorm_apply<<<g1d((long)M * D_HID), 256, 0, stream>>>(
            hout, M, D_HID, gn_w, gn_b, gn_ms, colsum, colsumsq);
    };

    // ---- cell branch (sage0..2) ----
    sage_layer(x_cell, N_CELL, D_FEAT, eic, E_cell, wt_of[5],  P(6),  wt_of[7],  aggc, hc0);
    sage_layer(hc0,    N_CELL, D_HID,  eic, E_cell, wt_of[8],  P(9),  wt_of[10], aggc, hc1);
    sage_layer(hc1,    N_CELL, D_HID,  eic, E_cell, wt_of[11], P(12), wt_of[13], aggc, hc0);
    // ---- tissue branch (sage3..5) ----
    sage_layer(x_tissue, N_TISSUE, D_FEAT, eit, E_tis, wt_of[14], P(15), wt_of[16], aggt, ht0);
    sage_layer(ht0,      N_TISSUE, D_HID,  eit, E_tis, wt_of[17], P(18), wt_of[19], aggt, ht1);
    sage_layer(ht1,      N_TISSUE, D_HID,  eit, E_tis, wt_of[20], P(21), wt_of[22], aggt, ht0);

    // ---- tokens ----
    build_tokens<<<g1d((long)TOKN), 256, 0, stream>>>(hc0, ht0, assign, P(59), P(58), X);

    // ---- transformer encoder x2 ----
    const int Mrow = N_CELL * N_TOK; // 98304
    for (int l = 0; l < 2; ++l) {
        int bI = 26 + l * 16;
        const float *bq = P(bI+1), *bk = P(bI+3), *bv = P(bI+5), *bo = P(bI+7);
        const float *l1w = P(bI+8), *l1b = P(bI+9);
        const float *b1 = P(bI+11), *b2 = P(bI+13);
        const float *l2w = P(bI+14), *l2b = P(bI+15);
        const _Float16 *Wq = wt_of[bI+0], *Wk = wt_of[bI+2], *Wv = wt_of[bI+4];
        const _Float16 *Wo = wt_of[bI+6], *W1 = wt_of[bI+10], *W2 = wt_of[bI+12];

        dim3 gg(Mrow / 64, D_HID / 64);
        gemm_wmma<false, 0><<<gg, 128, 0, stream>>>(X, D_HID, nullptr, 0, Wq, nullptr, bq, Qb, D_HID, Mrow, D_HID, D_HID);
        gemm_wmma<false, 0><<<gg, 128, 0, stream>>>(X, D_HID, nullptr, 0, Wk, nullptr, bk, Kb, D_HID, Mrow, D_HID, D_HID);
        gemm_wmma<false, 0><<<gg, 128, 0, stream>>>(X, D_HID, nullptr, 0, Wv, nullptr, bv, Vb, D_HID, Mrow, D_HID, D_HID);
        attn3<<<g1d((long)N_CELL * N_HEAD * 32), 256, 0, stream>>>(Qb, Kb, Vb, Tb);
        gemm_wmma<false, 0><<<gg, 128, 0, stream>>>(Tb, D_HID, nullptr, 0, Wo, nullptr, bo, Qb, D_HID, Mrow, D_HID, D_HID);
        add_ln<<<g1d((long)Mrow * 32), 256, 0, stream>>>(X, Qb, l1w, l1b, Mrow);
        gemm_wmma<false, 1><<<gg, 128, 0, stream>>>(X, D_HID, nullptr, 0, W1, nullptr, b1, Tb, D_HID, Mrow, D_HID, D_HID);
        gemm_wmma<false, 0><<<gg, 128, 0, stream>>>(Tb, D_HID, nullptr, 0, W2, nullptr, b2, Qb, D_HID, Mrow, D_HID, D_HID);
        add_ln<<<g1d((long)Mrow * 32), 256, 0, stream>>>(X, Qb, l2w, l2b, Mrow);
    }

    // ---- head: attention scores ----
    // xcls = X rows with stride 768 (CLS token of each cell)
    gemm_wmma<false, 2><<<dim3(N_CELL / 64, 128 / 64), 128, 0, stream>>>(
        X, N_TOK * D_HID, nullptr, 0, wt_of[60] /*A1*/, nullptr, P(61) /*a1*/,
        Tb, 128, N_CELL, 128, D_HID);
    att_dot<<<g1d(N_CELL), 256, 0, stream>>>(Tb, P(62) /*A2*/, P(63) /*a2*/, attraw);
    softmax_col<<<1, 256, 0, stream>>>(attraw, N_CELL, outAtt);

    // ---- head: graph embedding -> logits ----
    zerof<<<1, 256, 0, stream>>>(colsum, 256);
    zerof<<<1, 256, 0, stream>>>(colsumsq, 256);
    col_accum<<<g1d(N_CELL, 128).x, 256, 0, stream>>>(X, N_TOK * D_HID, N_CELL, D_HID, colsum, colsumsq);
    head_tail<<<1, 256, 0, stream>>>(colsum, P(64), P(65), P(66), P(67), P(68), P(69), logitsb);

    // ---- pack outputs ----
    write_out<<<g1d(CLASSES + (long)2 * E_cell), 256, 0, stream>>>(logitsb, eic, out, 2 * E_cell);
}
